// DglHGTGRUConv_39367670235356
// MI455X (gfx1250) — matrix operand order repaired and checked by reading, not compiled, not run on hardware
//
#include <hip/hip_runtime.h>
#include <hip/hip_bf16.h>
#include <math.h>

#define NNODE 100000
#define INDIM 128
#define OUTD  128
#define NHEAD 8
#define DK    16
#define NREL  3
#define NEDGE 400000
#define TSTEP 2
#define GK    128   // K dimension shared by all GEMMs in this model

typedef __attribute__((ext_vector_type(16))) __bf16 v16bf;
typedef __attribute__((ext_vector_type(8)))  float  v8f;

union BFrag { v16bf v; unsigned u[8]; };

__device__ __forceinline__ unsigned short f2bf(float f) {
  unsigned u = __float_as_uint(f);
  u += 0x7fffu + ((u >> 16) & 1u);   // round-to-nearest-even
  return (unsigned short)(u >> 16);
}

#if defined(__has_builtin)
#if __has_builtin(__builtin_amdgcn_cvt_pk_bf16_f32)
#define HAVE_CVT_PK_BF16 1
#endif
#endif

__device__ __forceinline__ unsigned pack_bf16(float lo, float hi) {
#ifdef HAVE_CVT_PK_BF16
  auto p = __builtin_amdgcn_cvt_pk_bf16_f32(lo, hi);   // v_cvt_pk_bf16_f32
  union { decltype(p) v; unsigned u; } cv;
  cv.v = p;
  return cv.u;
#else
  return (unsigned)f2bf(lo) | ((unsigned)f2bf(hi) << 16);
#endif
}

// order-preserving float<->uint mapping so atomicMax(u32) == float max (incl. negatives)
__device__ __forceinline__ unsigned ordf(float f) {
  unsigned u = __float_as_uint(f);
  return (u & 0x80000000u) ? ~u : (u | 0x80000000u);
}
__device__ __forceinline__ float unordf(unsigned u) {
  return __uint_as_float((u & 0x80000000u) ? (u ^ 0x80000000u) : ~u);
}

// ---------------------------------------------------------------------------
// C[M,P] = A[M,128] * W[P,128]^T + bias[P]  via v_wmma_f32_16x16x32_bf16
// one wave = one 16x16 output tile; block = 8 waves = 128 rows; K unrolled (4 WMMAs)
// ---------------------------------------------------------------------------
__global__ void __launch_bounds__(256) gemm_bf16_wmma(
    const float* __restrict__ A, const float* __restrict__ W,
    const float* __restrict__ bias, float* __restrict__ C,
    int M, int P) {
  int wave = threadIdx.x >> 5;
  int lane = threadIdx.x & 31;
  int hi   = lane >> 4;
  int l15  = lane & 15;
  int rowBase = blockIdx.x * 128 + wave * 16;
  int col     = blockIdx.y * 16 + l15;
  int arow = rowBase + l15;
  if (arow >= M) arow = M - 1;                  // clamp: keep EXEC all-ones for WMMA
  const float2* Ap = (const float2*)(A + (size_t)arow * GK);
  const float2* Wp = (const float2*)(W + (size_t)col  * GK);
  v8f acc = {};
#pragma unroll
  for (int kc = 0; kc < GK; kc += 32) {
    BFrag a, b;
#pragma unroll
    for (int j = 0; j < 8; ++j) {
      // A 16x32 layout: VGPR j, lane-half hi -> K = (j<4?0:16) + hi*8 + (j&3)*2
      int ka = (kc + ((j & 4) << 2) + hi * 8 + ((j & 3) << 1)) >> 1;  // float2 index
      float2 af = Ap[ka];
      a.u[j] = pack_bf16(af.x, af.y);
      // B 32x16 layout: VGPR j, lane-half hi -> K = (j<4?0:16) + hi*8 + (j&3)*2 ; N = l15
      int kb = (kc + ((j & 4) << 2) + hi * 8 + ((j & 3) << 1)) >> 1;
      float2 bf = Wp[kb];
      b.u[j] = pack_bf16(bf.x, bf.y);
    }
    acc = __builtin_amdgcn_wmma_f32_16x16x32_bf16(false, a.v, false, b.v,
                                                  (short)0, acc, false, false);
  }
  float bb = bias[col];
  float* Cp = C + (size_t)(rowBase + hi * 8) * P + col;  // D layout: M = r + hi*8
  if (rowBase + 16 <= M) {          // wave-uniform fast path: full tile in range
#pragma unroll
    for (int r = 0; r < 8; ++r) Cp[(size_t)r * P] = acc[r] + bb;
  } else {
#pragma unroll
    for (int r = 0; r < 8; ++r) {
      int row = rowBase + hi * 8 + r;
      if (row < M) C[(size_t)row * P + col] = acc[r] + bb;
    }
  }
}

// ---------------------------------------------------------------------------
// kr[n,h,:] = k[n,h,:] @ att[h], vr[n,h,:] = v[n,h,:] @ msg[h]; mats cached in LDS
// ---------------------------------------------------------------------------
__global__ void __launch_bounds__(256) relmap_kernel(
    const float* __restrict__ k, const float* __restrict__ v,
    const float* __restrict__ att, const float* __restrict__ msg,
    float* __restrict__ kr, float* __restrict__ vr) {
  __shared__ float sA[NHEAD * DK * DK];
  __shared__ float sM[NHEAD * DK * DK];
  for (int i = threadIdx.x; i < NHEAD * DK * DK; i += blockDim.x) {
    sA[i] = att[i];
    sM[i] = msg[i];
  }
  __syncthreads();
  int gid = blockIdx.x * blockDim.x + threadIdx.x;
  if (gid >= NNODE * NHEAD) return;
  int node = gid >> 3, hh = gid & 7;
  const float* kp = k + (size_t)node * OUTD + hh * DK;
  const float* vp = v + (size_t)node * OUTD + hh * DK;
  float kin[DK], vin[DK];
#pragma unroll
  for (int d = 0; d < DK; ++d) { kin[d] = kp[d]; vin[d] = vp[d]; }
  float* kro = kr + (size_t)node * OUTD + hh * DK;
  float* vro = vr + (size_t)node * OUTD + hh * DK;
#pragma unroll
  for (int e = 0; e < DK; ++e) {
    float ak = 0.f, av = 0.f;
#pragma unroll
    for (int d = 0; d < DK; ++d) {
      ak += kin[d] * sA[hh * 256 + d * 16 + e];
      av += vin[d] * sM[hh * 256 + d * 16 + e];
    }
    kro[e] = ak;
    vro[e] = av;
  }
}

__global__ void zero_f(float* __restrict__ p, int n) {
  int g = blockIdx.x * blockDim.x + threadIdx.x;
  if (g < n) p[g] = 0.f;
}

__global__ void init_ms(unsigned* __restrict__ m, float* __restrict__ s, int n) {
  int g = blockIdx.x * blockDim.x + threadIdx.x;
  if (g < n) { m[g] = 0u; s[g] = 0.f; }   // 0 == ordered lower bound (below ord(-inf))
}

// SDDMM: score[e,h] = dot(q[dst], kr[src]) * pri[h] / sqrt(DK); segment max via ordered atomicMax
__global__ void score_kernel(
    const float* __restrict__ q, const float* __restrict__ kr,
    const int* __restrict__ src, const int* __restrict__ dst,
    const float* __restrict__ pri, float* __restrict__ score,
    unsigned* __restrict__ mbuf, int nE) {
  int gid = blockIdx.x * blockDim.x + threadIdx.x;
  if (gid >= nE * NHEAD) return;
  int e = gid >> 3, hh = gid & 7;
  int s = src[e], d = dst[e];
  const float* qp = q  + (size_t)d * OUTD + hh * DK;
  const float* kp = kr + (size_t)s * OUTD + hh * DK;
  float acc = 0.f;
#pragma unroll
  for (int i = 0; i < DK; ++i) acc += qp[i] * kp[i];
  acc *= pri[hh] * 0.25f;  // 1/sqrt(16)
  score[gid] = acc;
  atomicMax(&mbuf[(size_t)d * NHEAD + hh], ordf(acc));
}

__global__ void exp_kernel(
    float* __restrict__ score, const int* __restrict__ dst,
    const unsigned* __restrict__ mbuf, float* __restrict__ sbuf, int nE) {
  int gid = blockIdx.x * blockDim.x + threadIdx.x;
  if (gid >= nE * NHEAD) return;
  int e = gid >> 3, hh = gid & 7;
  int d = dst[e];
  float mm = unordf(mbuf[(size_t)d * NHEAD + hh]);
  float ex = expf(score[gid] - mm);
  score[gid] = ex;
  atomicAdd(&sbuf[(size_t)d * NHEAD + hh], ex);
}

__global__ void agg_kernel(
    const float* __restrict__ score, const float* __restrict__ vr,
    const int* __restrict__ src, const int* __restrict__ dst,
    const float* __restrict__ sbuf, float* __restrict__ agg, int nE) {
  int gid = blockIdx.x * blockDim.x + threadIdx.x;
  if (gid >= nE * NHEAD) return;
  int e = gid >> 3, hh = gid & 7;
  int s = src[e], d = dst[e];
  float w = score[gid] / sbuf[(size_t)d * NHEAD + hh];
  const float* vp = vr + (size_t)s * OUTD + hh * DK;
  float* ap = agg + (size_t)d * OUTD + hh * DK;
#pragma unroll
  for (int i = 0; i < DK; ++i) atomicAdd(&ap[i], vp[i] * w);
}

__global__ void gelu_kernel(const float* __restrict__ a, float* __restrict__ t, int n) {
  int g = blockIdx.x * blockDim.x + threadIdx.x;
  if (g < n) {
    float x = a[g];
    t[g] = 0.5f * x * (1.f + erff(x * 0.70710678118654752f));  // exact GELU
  }
}

__global__ void gru_kernel(
    const float* __restrict__ gi, const float* __restrict__ gh,
    const float* __restrict__ hin, float* __restrict__ hout, int n) {
  int gid = blockIdx.x * blockDim.x + threadIdx.x;
  if (gid >= n * OUTD) return;
  int node = gid >> 7, c = gid & 127;
  const float* gp = gi + (size_t)node * (3 * OUTD);
  const float* hp = gh + (size_t)node * (3 * OUTD);
  float ir = gp[c], iz = gp[OUTD + c], in_ = gp[2 * OUTD + c];
  float hr = hp[c], hz = hp[OUTD + c], hn  = hp[2 * OUTD + c];
  float hold = hin[gid];
  float rg = 1.f / (1.f + expf(-(ir + hr)));
  float z  = 1.f / (1.f + expf(-(iz + hz)));
  float nn = tanhf(in_ + rg * hn);
  hout[gid] = (1.f - z) * nn + z * hold;
}

extern "C" void kernel_launch(void* const* d_in, const int* in_sizes, int n_in,
                              void* d_out, int out_size, void* d_ws, size_t ws_size,
                              hipStream_t stream) {
  (void)in_sizes; (void)n_in; (void)out_size; (void)ws_size;
  const float* h0      = (const float*)d_in[0];
  const int*   src     = (const int*)d_in[1];
  const int*   dst     = (const int*)d_in[2];
  const float* Wk = (const float*)d_in[3];  const float* bk = (const float*)d_in[4];
  const float* Wq = (const float*)d_in[5];  const float* bq = (const float*)d_in[6];
  const float* Wv = (const float*)d_in[7];  const float* bv = (const float*)d_in[8];
  const float* Wa = (const float*)d_in[9];  const float* ba = (const float*)d_in[10];
  const float* rel_pri = (const float*)d_in[11];
  const float* rel_att = (const float*)d_in[12];
  const float* rel_msg = (const float*)d_in[13];
  const float* W_ih = (const float*)d_in[14]; const float* b_ih = (const float*)d_in[15];
  const float* W_hh = (const float*)d_in[16]; const float* b_hh = (const float*)d_in[17];
  float* hout = (float*)d_out;

  float* ws = (float*)d_ws;
  const size_t NO = (size_t)NNODE * OUTD;
  // workspace layout (floats):
  //   [0,       NO)   kb_  : k                (later part of gi [N,384])
  //   [NO,    2*NO)   qb_  : q                (later part of gi)
  //   [2*NO,  3*NO)   vb_  : v                (later part of gi)
  //   [3*NO,  6*NO)   ghb  : gh [N,384]
  //   [6*NO,  7*NO)   krb  : kr               (later t = gelu(agg))
  //   [7*NO,  8*NO)   vrb  : vr               (later x = t @ Wa^T)
  //   [8*NO,  9*NO)   agg
  //   [9*NO, ...)     scb [E,8], mb [N,8] u32, sb [N,8]
  float* kb_ = ws;
  float* qb_ = ws + NO;
  float* vb_ = ws + 2 * NO;
  float* gib = ws;                 // alias of k/q/v block (free after edge phase)
  float* ghb = ws + 3 * NO;
  float* krb = ws + 6 * NO;
  float* vrb = ws + 7 * NO;
  float* agg = ws + 8 * NO;
  float* scb = ws + 9 * NO;
  unsigned* mb = (unsigned*)(ws + 9 * NO + (size_t)NEDGE * NHEAD);
  float* sb = ws + 9 * NO + (size_t)NEDGE * NHEAD + (size_t)NNODE * NHEAD;
  float* tb = krb;
  float* xb = vrb;

  dim3 ggemm((NNODE + 127) / 128, OUTD / 16);
  dim3 ggemm3((NNODE + 127) / 128, (3 * OUTD) / 16);
  int nEH = NEDGE * NHEAD;
  int nNH = NNODE * NHEAD;
  int nNO = NNODE * OUTD;

  const float* hcur = h0;
  for (int t = 0; t < TSTEP; ++t) {
    gemm_bf16_wmma<<<ggemm, 256, 0, stream>>>(hcur, Wk, bk, kb_, NNODE, OUTD);
    gemm_bf16_wmma<<<ggemm, 256, 0, stream>>>(hcur, Wq, bq, qb_, NNODE, OUTD);
    gemm_bf16_wmma<<<ggemm, 256, 0, stream>>>(hcur, Wv, bv, vb_, NNODE, OUTD);
    zero_f<<<(nNO + 255) / 256, 256, 0, stream>>>(agg, nNO);
    for (int r = 0; r < NREL; ++r) {
      relmap_kernel<<<(nNH + 255) / 256, 256, 0, stream>>>(
          kb_, vb_, rel_att + (size_t)r * NHEAD * DK * DK,
          rel_msg + (size_t)r * NHEAD * DK * DK, krb, vrb);
      init_ms<<<(nNH + 255) / 256, 256, 0, stream>>>(mb, sb, nNH);
      score_kernel<<<(nEH + 255) / 256, 256, 0, stream>>>(
          qb_, krb, src + (size_t)r * NEDGE, dst + (size_t)r * NEDGE,
          rel_pri + r * NHEAD, scb, mb, NEDGE);
      exp_kernel<<<(nEH + 255) / 256, 256, 0, stream>>>(
          scb, dst + (size_t)r * NEDGE, mb, sb, NEDGE);
      agg_kernel<<<(nEH + 255) / 256, 256, 0, stream>>>(
          scb, vrb, src + (size_t)r * NEDGE, dst + (size_t)r * NEDGE, sb, agg, NEDGE);
    }
    gelu_kernel<<<(nNO + 255) / 256, 256, 0, stream>>>(agg, tb, nNO);
    gemm_bf16_wmma<<<ggemm, 256, 0, stream>>>(tb, Wa, ba, xb, NNODE, OUTD);
    gemm_bf16_wmma<<<ggemm3, 256, 0, stream>>>(xb, W_ih, b_ih, gib, NNODE, 3 * OUTD);
    gemm_bf16_wmma<<<ggemm3, 256, 0, stream>>>(hcur, W_hh, b_hh, ghb, NNODE, 3 * OUTD);
    gru_kernel<<<(nNO + 255) / 256, 256, 0, stream>>>(gib, ghb, hcur, hout, NNODE);
    hcur = hout;
  }
}
